// MultiHeadLatentAttention_8589934592341
// MI455X (gfx1250) — matrix operand order repaired and checked by reading, not compiled, not run on hardware
//
#include <hip/hip_runtime.h>

// ---------------------------------------------------------------------------
// MLA forward for MI455X (gfx1250, wave32, WMMA).
// Pipeline: cast->bf16, 5 WMMA GEMMs, RoPE+relayout, flash attention (WMMA),
// final WMMA GEMM -> f32 output.
// Round 3: fix async-to-LDS builtin pointer types (v4i AS1 / AS3 pointers).
// ---------------------------------------------------------------------------

typedef __attribute__((ext_vector_type(16))) __bf16 v16bf;
typedef __attribute__((ext_vector_type(8)))  float  v8f;

union FragB {
    v16bf  v;
    uint4  u[2];
    __bf16 h[16];
};

// Problem constants
constexpr int Bq   = 2;
constexpr int Sq   = 2048;
constexpr int HIDq = 2048;
constexpr int NHq  = 16;
constexpr int HDq  = 128;
constexpr int KVCq = 512;
constexpr int QCq  = 1536;
constexpr int Mq   = Bq * Sq;          // 4096 rows for all GEMMs
constexpr int NHHD = NHq * HDq;        // 2048

// --- CDNA5 async global->LDS path (guarded so either toolchain compiles) ---
#if defined(__AMDGCN__) && __has_builtin(__builtin_amdgcn_global_load_async_to_lds_b128)
#define HAVE_ASYNC_LDS 1
#else
#define HAVE_ASYNC_LDS 0
#endif

#if HAVE_ASYNC_LDS
typedef int v4i_vs __attribute__((vector_size(4 * sizeof(int))));
typedef __attribute__((address_space(1))) v4i_vs* gvec_p;   // global <4 x i32>*
typedef __attribute__((address_space(3))) v4i_vs* lvec_p;   // LDS    <4 x i32>*
#define AS_GLOBAL(p) ((gvec_p)(p))
#define AS_LDS(p)    ((lvec_p)(p))
#endif

__device__ __forceinline__ void wait_async0() {
#if HAVE_ASYNC_LDS
#if __has_builtin(__builtin_amdgcn_s_wait_asynccnt)
    __builtin_amdgcn_s_wait_asynccnt(0);
#else
    asm volatile("s_wait_asynccnt 0x0" ::: "memory");
#endif
#endif
}

// ---------------------------------------------------------------------------
// f32 -> bf16 cast (vectorized by 4)
// ---------------------------------------------------------------------------
__global__ void cast_f32_to_bf16(const float* __restrict__ in,
                                 __bf16* __restrict__ out, int n4) {
    int i = blockIdx.x * blockDim.x + threadIdx.x;
    if (i >= n4) return;
    float4 f = ((const float4*)in)[i];
    out[4 * i + 0] = (__bf16)f.x;
    out[4 * i + 1] = (__bf16)f.y;
    out[4 * i + 2] = (__bf16)f.z;
    out[4 * i + 3] = (__bf16)f.w;
}

// ---------------------------------------------------------------------------
// Generic bf16 WMMA GEMM:  C[M,N] = A[M,K] @ B[K,N] + bias[N]
// Block = 256 threads (8 waves), block tile 128x128, wave tile 64x32
// (4x2 WMMA tiles), K-step 32. A staged row-major (async copy when
// available), B staged K-transposed so fragment reads are 16B ds_load_b128s.
// ---------------------------------------------------------------------------
__global__ __launch_bounds__(256) void gemm_bf16_wmma(
    const __bf16* __restrict__ A, const __bf16* __restrict__ Bm,
    const float* __restrict__ bias, void* __restrict__ Co,
    int M, int N, int K, int out_f32) {
    __shared__ __bf16 As[128][32];   // [m][k]  (8KB)
    __shared__ __bf16 Bt[128][32];   // [n][k]  (8KB)

    const int tid  = threadIdx.x;
    const int lane = tid & 31;
    const int wave = tid >> 5;      // 0..7
    const int wm   = wave >> 2;     // 0..1 -> row offset *64
    const int wn   = wave & 3;      // 0..3 -> col offset *32
    const int lo   = lane & 15;
    const int hi   = lane >> 4;
    const int mBlk = blockIdx.y * 128;
    const int nBlk = blockIdx.x * 128;

    v8f acc[4][2];
#pragma unroll
    for (int a = 0; a < 4; ++a)
#pragma unroll
        for (int b = 0; b < 2; ++b)
#pragma unroll
            for (int r = 0; r < 8; ++r) acc[a][b][r] = 0.0f;

    const int arow = tid >> 1, acol = (tid & 1) * 16;   // A fill: 16 elems/thread
    const int brow = tid >> 3, bcol = (tid & 7) * 16;   // B fill: 16 elems/thread

    for (int k0 = 0; k0 < K; k0 += 32) {
        // --- stage A tile (128x32) ---
        const __bf16* ag = A + (size_t)(mBlk + arow) * K + k0 + acol;
#if HAVE_ASYNC_LDS
        __builtin_amdgcn_global_load_async_to_lds_b128(
            AS_GLOBAL(ag), AS_LDS(&As[arow][acol]), 0, 0);
        __builtin_amdgcn_global_load_async_to_lds_b128(
            AS_GLOBAL(ag + 8), AS_LDS(&As[arow][acol + 8]), 0, 0);
#else
        *(uint4*)&As[arow][acol]     = *(const uint4*)ag;
        *(uint4*)&As[arow][acol + 8] = *(const uint4*)(ag + 8);
#endif
        // --- stage B tile (32x128) transposed into Bt[n][k] ---
        FragB tb;
        const __bf16* bg = Bm + (size_t)(k0 + brow) * N + nBlk + bcol;
        tb.u[0] = *(const uint4*)bg;
        tb.u[1] = *(const uint4*)(bg + 8);
#pragma unroll
        for (int j = 0; j < 16; ++j) Bt[bcol + j][brow] = tb.h[j];

        if (k0 + 32 < K) {
            __builtin_prefetch(A + (size_t)(mBlk + arow) * K + k0 + 32 + acol, 0, 1);
            __builtin_prefetch(Bm + (size_t)(k0 + 32 + brow) * N + nBlk + bcol, 0, 1);
        }
        wait_async0();
        __syncthreads();

        // --- fragments (per ISA 16-bit A 16x32 / B 32x16 layouts) ---
        const int kb  = hi * 8;    // A: lanes 0-15 -> K 0..7/16..23, 16-31 -> 8..15/24..31
        const int kb2 = hi * 8;    // B: lanes 0-15 -> K 0..7/16..23, 16-31 -> 8..15/24..31
        FragB af[4], bfr[2];
#pragma unroll
        for (int t = 0; t < 4; ++t) {
            int m = wm * 64 + t * 16 + lo;
            af[t].u[0] = *(const uint4*)&As[m][kb];
            af[t].u[1] = *(const uint4*)&As[m][16 + kb];
        }
#pragma unroll
        for (int t = 0; t < 2; ++t) {
            int n = wn * 32 + t * 16 + lo;
            bfr[t].u[0] = *(const uint4*)&Bt[n][kb2];
            bfr[t].u[1] = *(const uint4*)&Bt[n][kb2 + 16];
        }
#pragma unroll
        for (int tm = 0; tm < 4; ++tm)
#pragma unroll
            for (int tn = 0; tn < 2; ++tn)
                acc[tm][tn] = __builtin_amdgcn_wmma_f32_16x16x32_bf16(
                    false, af[tm].v, false, bfr[tn].v, (short)0, acc[tm][tn],
                    false, false);
        __syncthreads();
    }

    // --- epilogue: bias + store (C layout: lane=col, vgpr r=row r+8*hi) ---
#pragma unroll
    for (int tn = 0; tn < 2; ++tn) {
        int ng   = nBlk + wn * 32 + tn * 16 + lo;
        float bv = bias[ng];
#pragma unroll
        for (int tm = 0; tm < 4; ++tm) {
#pragma unroll
            for (int r = 0; r < 8; ++r) {
                int mg  = mBlk + wm * 64 + tm * 16 + r + 8 * hi;
                float v = acc[tm][tn][r] + bv;
                if (out_f32)
                    ((float*)Co)[(size_t)mg * N + ng] = v;
                else
                    ((__bf16*)Co)[(size_t)mg * N + ng] = (__bf16)v;
            }
        }
    }
}

// ---------------------------------------------------------------------------
// RoPE (interleaved pairs) fused with [B,S,H,D] -> [B,H,S,D] relayout.
// One thread per (b,h,s,pair).
// ---------------------------------------------------------------------------
__global__ void rope_relayout(const __bf16* __restrict__ in,
                              __bf16* __restrict__ out, int apply) {
    int idx = blockIdx.x * blockDim.x + threadIdx.x;
    if (idx >= Bq * NHq * Sq * (HDq / 2)) return;
    int i = idx & 63;             // pair index 0..63
    int s = (idx >> 6) & (Sq - 1);
    int h = (idx >> 17) & (NHq - 1);
    int b = idx >> 21;

    size_t io = ((size_t)(b * Sq + s)) * NHHD + h * HDq + 2 * i;
    size_t oo = ((size_t)((b * NHq + h) * Sq + s)) * HDq + 2 * i;

    float x1 = (float)in[io], x2 = (float)in[io + 1];
    float o1 = x1, o2 = x2;
    if (apply) {
        // inv_freq = 10000^(-2i/128); ln(10000) = 9.210340371976184
        float inv = __expf(-(float)(2 * i) * (9.210340371976184f / 128.0f));
        float ang = (float)s * inv;
        float sn, cs;
        __sincosf(ang, &sn, &cs);
        o1 = x1 * cs - x2 * sn;
        o2 = x1 * sn + x2 * cs;
    }
    out[oo]     = (__bf16)o1;
    out[oo + 1] = (__bf16)o2;
}

// ---------------------------------------------------------------------------
// Flash attention (causal), WMMA bf16, f32 accumulators.
// Grid (S/64, NH, B); block = 128 threads (4 waves), each wave owns 16 q rows.
// Q,K in [B,H,S,D]; V and O in [B*S, H*D]. K tile staged with async copies.
// ---------------------------------------------------------------------------
__global__ __launch_bounds__(128) void mla_flash_attn(
    const __bf16* __restrict__ Q, const __bf16* __restrict__ Kc,
    const __bf16* __restrict__ V, __bf16* __restrict__ O) {
    __shared__ __bf16 Ks[64][128];      // key tile, row-major [key][d]   (16KB)
    __shared__ __bf16 Vt[128][64];      // value tile transposed [d][key] (16KB)
    __shared__ __bf16 Ps[4][16][64];    // per-wave P staging              (8KB)

    const int tid  = threadIdx.x;
    const int lane = tid & 31;
    const int wave = tid >> 5;
    const int lo   = lane & 15;
    const int hi   = lane >> 4;
    const int qt = blockIdx.x, h = blockIdx.y, b = blockIdx.z;
    const int q0 = qt * 64 + wave * 16;

    // Load Q fragments for this wave's 16 rows (4 K-steps over d=128).
    FragB qf[4];
    {
        const size_t qbase = ((size_t)((b * NHq + h) * Sq) + q0 + lo) * HDq;
        const int kb = hi * 8;
#pragma unroll
        for (int ks = 0; ks < 4; ++ks) {
            qf[ks].u[0] = *(const uint4*)(Q + qbase + ks * 32 + kb);
            qf[ks].u[1] = *(const uint4*)(Q + qbase + ks * 32 + 16 + kb);
        }
    }

    v8f oacc[8];
#pragma unroll
    for (int t = 0; t < 8; ++t)
#pragma unroll
        for (int r = 0; r < 8; ++r) oacc[t][r] = 0.0f;
    float mrun[8], lrun[8];
#pragma unroll
    for (int r = 0; r < 8; ++r) { mrun[r] = -1e30f; lrun[r] = 0.0f; }

    const float scale = 0.08838834764831845f;  // 1/sqrt(128)

    for (int kt = 0; kt <= qt; ++kt) {
        // --- cooperative fill of Ks (async copy) and Vt (transposed) ---
        {
            int row = tid >> 1, d0 = (tid & 1) * 64;
            const __bf16* kg =
                Kc + ((size_t)((b * NHq + h) * Sq) + kt * 64 + row) * HDq + d0;
#if HAVE_ASYNC_LDS
#pragma unroll
            for (int j = 0; j < 8; ++j)
                __builtin_amdgcn_global_load_async_to_lds_b128(
                    AS_GLOBAL(kg + j * 8), AS_LDS(&Ks[row][d0 + j * 8]), 0, 0);
#else
#pragma unroll
            for (int j = 0; j < 8; ++j)
                *(uint4*)&Ks[row][d0 + j * 8] = *(const uint4*)(kg + j * 8);
#endif
            const __bf16* vg =
                V + ((size_t)(b * Sq) + kt * 64 + row) * NHHD + h * HDq + d0;
#pragma unroll
            for (int j = 0; j < 8; ++j) {
                FragB tv;
                tv.u[0] = *(const uint4*)(vg + j * 8);
#pragma unroll
                for (int e = 0; e < 8; ++e) Vt[d0 + j * 8 + e][row] = tv.h[e];
            }
        }
        wait_async0();
        __syncthreads();

        // --- scores: 4 tiles of 16q x 16k, each 4 WMMAs over d ---
        v8f st[4];
#pragma unroll
        for (int nt = 0; nt < 4; ++nt) {
            v8f sa;
#pragma unroll
            for (int r = 0; r < 8; ++r) sa[r] = 0.0f;
            int n = nt * 16 + lo;  // key within tile (B-matrix column)
#pragma unroll
            for (int ks = 0; ks < 4; ++ks) {
                FragB bfrag;
                int off      = ks * 32 + hi * 8;
                bfrag.u[0]   = *(const uint4*)&Ks[n][off];
                bfrag.u[1]   = *(const uint4*)&Ks[n][off + 16];
                sa = __builtin_amdgcn_wmma_f32_16x16x32_bf16(
                    false, qf[ks].v, false, bfrag.v, (short)0, sa, false, false);
            }
#pragma unroll
            for (int r = 0; r < 8; ++r) sa[r] *= scale;
            if (kt == qt) {
                int keyg = kt * 64 + n;
#pragma unroll
                for (int r = 0; r < 8; ++r) {
                    int qrow = q0 + r + 8 * hi;
                    if (keyg > qrow) sa[r] = -1e30f;
                }
            }
            st[nt] = sa;
        }

        // --- online softmax (row reductions across the 16-lane column group) ---
        float fr[8], mnew[8], rs[8];
#pragma unroll
        for (int r = 0; r < 8; ++r) {
            float tm = fmaxf(fmaxf(st[0][r], st[1][r]), fmaxf(st[2][r], st[3][r]));
#pragma unroll
            for (int mk = 1; mk < 16; mk <<= 1)
                tm = fmaxf(tm, __shfl_xor(tm, mk, 32));
            mnew[r] = fmaxf(mrun[r], tm);
            fr[r]   = __expf(mrun[r] - mnew[r]);
            mrun[r] = mnew[r];
            rs[r]   = 0.0f;
        }
#pragma unroll
        for (int nt = 0; nt < 4; ++nt) {
#pragma unroll
            for (int r = 0; r < 8; ++r) {
                float p = __expf(st[nt][r] - mnew[r]);
                rs[r] += p;
                Ps[wave][r + 8 * hi][nt * 16 + lo] = (__bf16)p;
            }
        }
#pragma unroll
        for (int r = 0; r < 8; ++r) {
            float ssum = rs[r];
#pragma unroll
            for (int mk = 1; mk < 16; mk <<= 1) ssum += __shfl_xor(ssum, mk, 32);
            lrun[r] = lrun[r] * fr[r] + ssum;
#pragma unroll
            for (int t = 0; t < 8; ++t) oacc[t][r] *= fr[r];
        }
        __builtin_amdgcn_wave_barrier();  // wave-private LDS RAW ordering hint

        // --- P @ V: 8 d-tiles x 2 key-steps = 16 WMMAs into oacc ---
#pragma unroll
        for (int ks2 = 0; ks2 < 2; ++ks2) {
            FragB pf;
            int kbp  = hi * 8;
            pf.u[0] = *(const uint4*)&Ps[wave][lo][ks2 * 32 + kbp];
            pf.u[1] = *(const uint4*)&Ps[wave][lo][ks2 * 32 + 16 + kbp];
#pragma unroll
            for (int nt2 = 0; nt2 < 8; ++nt2) {
                FragB vf;
                int dcol = nt2 * 16 + lo;
                int koff = ks2 * 32 + hi * 8;
                vf.u[0] = *(const uint4*)&Vt[dcol][koff];
                vf.u[1] = *(const uint4*)&Vt[dcol][koff + 16];
                oacc[nt2] = __builtin_amdgcn_wmma_f32_16x16x32_bf16(
                    false, pf.v, false, vf.v, (short)0, oacc[nt2], false, false);
            }
        }
        __syncthreads();
    }

    // --- normalize and store O in [B*S, H*D] (ready for the W_o GEMM) ---
#pragma unroll
    for (int nt2 = 0; nt2 < 8; ++nt2) {
        int dcol = nt2 * 16 + lo;
#pragma unroll
        for (int r = 0; r < 8; ++r) {
            int qrow = q0 + r + 8 * hi;
            float v  = oacc[nt2][r] / lrun[r];
            O[((size_t)(b * Sq) + qrow) * NHHD + h * HDq + dcol] = (__bf16)v;
        }
    }
}

// ---------------------------------------------------------------------------
// Host launcher
// ---------------------------------------------------------------------------
extern "C" void kernel_launch(void* const* d_in, const int* in_sizes, int n_in,
                              void* d_out, int out_size, void* d_ws,
                              size_t ws_size, hipStream_t stream) {
    (void)in_sizes; (void)n_in; (void)out_size; (void)ws_size;

    const float* x     = (const float*)d_in[0];
    // d_in[1] = attention_mask (causal; handled analytically)
    const float* W_dkv = (const float*)d_in[2];
    const float* b_dkv = (const float*)d_in[3];
    const float* W_uk  = (const float*)d_in[4];
    const float* b_uk  = (const float*)d_in[5];
    const float* W_uv  = (const float*)d_in[6];
    const float* b_uv  = (const float*)d_in[7];
    const float* W_dq  = (const float*)d_in[8];
    const float* b_dq  = (const float*)d_in[9];
    const float* W_uq  = (const float*)d_in[10];
    const float* b_uq  = (const float*)d_in[11];
    const float* W_o   = (const float*)d_in[12];
    const float* b_o   = (const float*)d_in[13];

    char*  ws  = (char*)d_ws;
    size_t off = 0;
    auto alloc_bf16 = [&](size_t elems) -> __bf16* {
        __bf16* p = (__bf16*)(ws + off);
        off += (elems * sizeof(__bf16) + 255) & ~(size_t)255;
        return p;
    };

    __bf16* xb    = alloc_bf16((size_t)Mq * HIDq);
    __bf16* wdkv  = alloc_bf16((size_t)HIDq * KVCq);
    __bf16* wuk   = alloc_bf16((size_t)KVCq * NHHD);
    __bf16* wuv   = alloc_bf16((size_t)KVCq * NHHD);
    __bf16* wdq   = alloc_bf16((size_t)HIDq * QCq);
    __bf16* wuq   = alloc_bf16((size_t)QCq * NHHD);
    __bf16* wob   = alloc_bf16((size_t)NHHD * HIDq);
    __bf16* kvb   = alloc_bf16((size_t)Mq * KVCq);
    __bf16* kfull = alloc_bf16((size_t)Mq * NHHD);
    __bf16* vb    = alloc_bf16((size_t)Mq * NHHD);
    __bf16* qcb   = alloc_bf16((size_t)Mq * QCq);
    __bf16* qfull = alloc_bf16((size_t)Mq * NHHD);
    __bf16* qr    = alloc_bf16((size_t)Mq * NHHD);
    __bf16* kr    = alloc_bf16((size_t)Mq * NHHD);
    __bf16* ob    = alloc_bf16((size_t)Mq * NHHD);

    auto cast = [&](const float* src, __bf16* dst, size_t n) {
        int n4 = (int)(n / 4);
        cast_f32_to_bf16<<<(n4 + 255) / 256, 256, 0, stream>>>(src, dst, n4);
    };
    auto gemm = [&](const __bf16* A, const __bf16* Bm, const float* bias,
                    void* Co, int M_, int N_, int K_, int f32out) {
        dim3 g(N_ / 128, M_ / 128);
        gemm_bf16_wmma<<<g, 256, 0, stream>>>(A, Bm, bias, Co, M_, N_, K_, f32out);
    };

    // 1) casts
    cast(x, xb, (size_t)Mq * HIDq);
    cast(W_dkv, wdkv, (size_t)HIDq * KVCq);
    cast(W_uk, wuk, (size_t)KVCq * NHHD);
    cast(W_uv, wuv, (size_t)KVCq * NHHD);
    cast(W_dq, wdq, (size_t)HIDq * QCq);
    cast(W_uq, wuq, (size_t)QCq * NHHD);
    cast(W_o, wob, (size_t)NHHD * HIDq);

    // 2) projections
    gemm(xb, wdkv, b_dkv, kvb, Mq, KVCq, HIDq, 0);     // kv  = x @ W_dkv + b
    gemm(kvb, wuk, b_uk, kfull, Mq, NHHD, KVCq, 0);    // k   = kv @ W_uk + b
    gemm(kvb, wuv, b_uv, vb, Mq, NHHD, KVCq, 0);       // v   = kv @ W_uv + b
    gemm(xb, wdq, b_dq, qcb, Mq, QCq, HIDq, 0);        // q_c = x @ W_dq + b
    gemm(qcb, wuq, b_uq, qfull, Mq, NHHD, QCq, 0);     // q   = q_c @ W_uq + b

    // 3) RoPE + relayout to [B,H,S,D]
    int nrope = Bq * NHq * Sq * (HDq / 2);
    rope_relayout<<<(nrope + 255) / 256, 256, 0, stream>>>(qfull, qr, 1);
    rope_relayout<<<(nrope + 255) / 256, 256, 0, stream>>>(kfull, kr, 1);

    // 4) causal flash attention -> ob in [B*S, H*D]
    mla_flash_attn<<<dim3(Sq / 64, NHq, Bq), 128, 0, stream>>>(qr, kr, vb, ob);

    // 5) output projection (f32 result straight to d_out)
    gemm(ob, wob, b_o, d_out, Mq, HIDq, NHHD, 1);
}
